// TiNSR_46720654246167
// MI455X (gfx1250) — matrix operand order repaired and checked by reading, not compiled, not run on hardware
//
#include <hip/hip_runtime.h>
#include <hip/hip_bf16.h>

typedef _Float16 f16;
typedef __attribute__((ext_vector_type(16))) _Float16 v16h;
typedef __attribute__((ext_vector_type(8)))  _Float16 h8;
typedef __attribute__((ext_vector_type(8)))  float    v8f;

#define N_RAYS    16384
#define N_SAMPLES 524288
#define STEP      0.005f
#define PI_F      3.14159265358979f

// ---------------- WMMA helpers (gfx1250, wave32) ----------------
__device__ __forceinline__ v8f wmma32(v16h a, v16h b, v8f c) {
  return __builtin_amdgcn_wmma_f32_16x16x32_f16(false, a, false, b, (short)0, c, false, false);
}

// A fragment: 16x32 f16, row-major source, rows r=lane&15, K-swizzled per ISA:
// lo lanes: K = kb+[0..7] and kb+[16..23]; hi lanes: kb+[8..15], kb+[24..31]
__device__ __forceinline__ v16h frag_a(const f16* base, int ld, int kb) {
  int lane = threadIdx.x & 31;
  int r = lane & 15, hi = lane >> 4;
  const f16* q = base + (size_t)r * ld + kb + hi * 8;
  h8 a = *(const h8*)q;
  h8 b = *(const h8*)(q + 16);
  return __builtin_shufflevector(a, b, 0,1,2,3,4,5,6,7,8,9,10,11,12,13,14,15);
}

// B fragment: 32x16 f16 from transposed weight WT[N][Kpad] row-major.
// lane n=lane&15 -> column n; lo lanes: K = kb+[0..7], kb+[16..23]; hi lanes: kb+[8..15], kb+[24..31]
__device__ __forceinline__ v16h frag_b(const f16* wt, int ldk, int kb) {
  int lane = threadIdx.x & 31;
  int n = lane & 15, hi = lane >> 4;
  const f16* q = wt + (size_t)n * ldk + kb + hi * 8;
  h8 a = *(const h8*)q;
  h8 b = *(const h8*)(q + 16);
  return __builtin_shufflevector(a, b, 0,1,2,3,4,5,6,7,8,9,10,11,12,13,14,15);
}

// Masked A fragment for backward: g_z1 = (h1>0) ? W2col0[k] : 0, built on the fly.
__device__ __forceinline__ v16h frag_a_mask(const f16* hbase, int ld, const f16* w2c, int kb) {
  int lane = threadIdx.x & 31;
  int r = lane & 15, hi = lane >> 4;
  const f16* q  = hbase + (size_t)r * ld + kb + hi * 8;
  const f16* wq = w2c + kb + hi * 8;
  v16h out;
#pragma unroll
  for (int j = 0; j < 8; ++j) {
    out[j]     = (q[j]      > (f16)0) ? wq[j]      : (f16)0;
    out[j + 8] = (q[j + 16] > (f16)0) ? wq[j + 16] : (f16)0;
  }
  return out;
}

// ---------------- weight prep: f32 -> f16, transposes, padding ----------------
__global__ void k_wprep(const float* __restrict__ Wg0, const float* __restrict__ Wg1,
                        const float* __restrict__ Wg2, const float* __restrict__ Wc0,
                        f16* __restrict__ W0T, f16* __restrict__ W1T, f16* __restrict__ W1f,
                        f16* __restrict__ W2T, f16* __restrict__ W0f, f16* __restrict__ Wc0T) {
  int i = blockIdx.x * 256 + threadIdx.x;
  if (i < 65536) {                       // Wg1 (256x256): straight f16 + transpose
    int k = i >> 8, n = i & 255;
    f16 w = (f16)Wg1[i];
    W1f[i] = w;                          // row-major (B^T for backward g_h0)
    W1T[(size_t)n * 256 + k] = w;        // transposed (B^T for forward L1)
  }
  if (i < 16384) {
    { int n = i >> 6, k = i & 63;        // W0T [256][64], K padded 39->64
      W0T[i] = (k < 39) ? (f16)Wg0[(size_t)k * 256 + n] : (f16)0; }
    { int r = i >> 8, c = i & 255;       // W0f [64][256], rows padded 39->64
      W0f[i] = (r < 39) ? (f16)Wg0[(size_t)r * 256 + c] : (f16)0; }
  }
  if (i < 4096) {                        // W2T [16][256]
    int n = i >> 8, k = i & 255;
    W2T[i] = (f16)Wg2[(size_t)k * 16 + n];
  }
  if (i < 8192) {                        // Wc0T [256][32], K padded 25->32
    int n = i >> 5, k = i & 31;
    Wc0T[i] = (k < 25) ? (f16)Wc0[(size_t)k * 256 + n] : (f16)0;
  }
}

// ---------------- per-sample prep: gather rays, normalize dir, points ----------------
__global__ void k_prep(const float* __restrict__ rays_o, const float* __restrict__ rays_d,
                       const int* __restrict__ ridx, const float* __restrict__ t_st,
                       float* __restrict__ points, float* __restrict__ dirs) {
  int s = blockIdx.x * 256 + threadIdx.x;
  if (s >= N_SAMPLES) return;
  int r = ridx[s];
  float dx = rays_d[r * 3 + 0], dy = rays_d[r * 3 + 1], dz = rays_d[r * 3 + 2];
  float inv = 1.0f / (sqrtf(dx * dx + dy * dy + dz * dz) + 1e-10f);
  dx *= inv; dy *= inv; dz *= inv;
  float mid = t_st[s] + STEP * 0.5f;
  points[s * 3 + 0] = rays_o[r * 3 + 0] + dx * mid;
  points[s * 3 + 1] = rays_o[r * 3 + 1] + dy * mid;
  points[s * 3 + 2] = rays_o[r * 3 + 2] + dz * mid;
  dirs[s * 3 + 0] = dx; dirs[s * 3 + 1] = dy; dirs[s * 3 + 2] = dz;
}

// ---------------- fused geometry MLP forward + VJP (WMMA) ----------------
// Tile: 32 samples/block, 256 threads = 8 waves.
// Wave w owns a 32-column slice (cols w*32..w*32+31) across BOTH 16-row tiles,
// so every B fragment (global) feeds 2 WMMAs (2x B reuse vs row-split mapping).
__global__ __launch_bounds__(256) void k_geom(
    const float* __restrict__ points,
    const f16* __restrict__ W0T, const float* __restrict__ bg0,
    const f16* __restrict__ W1T, const float* __restrict__ bg1,
    const f16* __restrict__ W2T, const float* __restrict__ bg2,
    const f16* __restrict__ W1f, const f16* __restrict__ W0f,
    const float* __restrict__ Wg2f32,
    float* __restrict__ feats, float* __restrict__ gradp) {
  const int MT = 32, XP = 72, HP = 272;
  __shared__ f16   H0s[32 * 272];
  __shared__ f16   H1s[32 * 272];
  __shared__ float GXs[32 * 64];     // also aliased as Xs (pos-enc tile) early on
  __shared__ f16   w2c[256];
  f16* Xs = (f16*)GXs;               // 32*72 f16 = 4608B <= 8192B

  int t = threadIdx.x;
  int s0 = blockIdx.x * MT;
  for (int i = t; i < 256; i += 256) w2c[i] = (f16)Wg2f32[(size_t)i * 16]; // Wg2[:,0]

  // positional encoding tile, zero-padded cols 39..71
  for (int idx = t; idx < MT * XP; idx += 256) {
    int r = idx / XP, c = idx % XP;
    const float* p = points + (size_t)(s0 + r) * 3;
    float v = 0.f;
    if (c < 3) v = p[c];
    else if (c < 39) {
      int q = c - 3, f = q / 6, rm = q % 6, d = rm % 3;
      float fr = PI_F * (float)(1 << f);
      float xb = p[d] * fr;
      v = (rm < 3) ? __sinf(xb) : __cosf(xb);
    }
    Xs[r * XP + c] = (f16)v;
  }
  __syncthreads();

  int wave = t >> 5;
  int lane = t & 31, cn = lane & 15, chi = lane >> 4;

  // ---- layer 0: [32x64] @ W0 -> h0 [32x256]
  {
    v16h xa[2][2];
#pragma unroll
    for (int mt = 0; mt < 2; ++mt)
#pragma unroll
      for (int k = 0; k < 2; ++k)
        xa[mt][k] = frag_a(Xs + mt * 16 * XP, XP, k * 32);
#pragma unroll
    for (int nt = 0; nt < 2; ++nt) {
      int nb = wave * 32 + nt * 16;
      v8f c0 = {}, c1 = {};
#pragma unroll
      for (int k = 0; k < 2; ++k) {
        v16h b = frag_b(W0T + (size_t)nb * 64, 64, k * 32);
        c0 = wmma32(xa[0][k], b, c0);
        c1 = wmma32(xa[1][k], b, c1);
      }
      float bb = bg0[nb + cn];
#pragma unroll
      for (int v = 0; v < 8; ++v) {
        H0s[(v + 8 * chi) * HP + nb + cn]      = (f16)fmaxf(c0[v] + bb, 0.f);
        H0s[(16 + v + 8 * chi) * HP + nb + cn] = (f16)fmaxf(c1[v] + bb, 0.f);
      }
    }
  }
  __syncthreads();

  // ---- layer 1: h0 @ W1 -> h1 [32x256]  (each B frag reused for 2 row tiles)
#pragma unroll
  for (int nt = 0; nt < 2; ++nt) {
    int nb = wave * 32 + nt * 16;
    v8f c0 = {}, c1 = {};
#pragma unroll
    for (int k = 0; k < 8; ++k) {
      v16h b = frag_b(W1T + (size_t)nb * 256, 256, k * 32);
      c0 = wmma32(frag_a(H0s, HP, k * 32), b, c0);
      c1 = wmma32(frag_a(H0s + 16 * HP, HP, k * 32), b, c1);
    }
    float bb = bg1[nb + cn];
#pragma unroll
    for (int v = 0; v < 8; ++v) {
      H1s[(v + 8 * chi) * HP + nb + cn]      = (f16)fmaxf(c0[v] + bb, 0.f);
      H1s[(16 + v + 8 * chi) * HP + nb + cn] = (f16)fmaxf(c1[v] + bb, 0.f);
    }
  }
  __syncthreads();

  // ---- layer 2 (waves 0,1): h1 @ W2 -> feats [32x16]
  if (wave < 2) {
    const f16* hb = H1s + wave * 16 * HP;
    v8f c = {};
#pragma unroll
    for (int k = 0; k < 8; ++k)
      c = wmma32(frag_a(hb, HP, k * 32), frag_b(W2T, 256, k * 32), c);
    float bb = bg2[cn];
#pragma unroll
    for (int v = 0; v < 8; ++v)
      feats[(size_t)(s0 + wave * 16 + v + 8 * chi) * 16 + cn] = c[v] + bb;
  }

  // ---- backward: g_z1 = (h1>0)?W2[:,0]:0 built on the fly; g_h0 = g_z1 @ W1^T
#pragma unroll
  for (int nt = 0; nt < 2; ++nt) {
    int nb = wave * 32 + nt * 16;
    v8f c0 = {}, c1 = {};
#pragma unroll
    for (int k = 0; k < 8; ++k) {
      v16h b = frag_b(W1f + (size_t)nb * 256, 256, k * 32);
      c0 = wmma32(frag_a_mask(H1s, HP, w2c, k * 32), b, c0);
      c1 = wmma32(frag_a_mask(H1s + 16 * HP, HP, w2c, k * 32), b, c1);
    }
    // g_z0 = (h0>0) ? g_h0 : 0, written in place over H0s (own cols only)
#pragma unroll
    for (int v = 0; v < 8; ++v) {
      int m0 = v + 8 * chi, m1 = 16 + v + 8 * chi, n = nb + cn;
      float g0v = (H0s[m0 * HP + n] > (f16)0) ? c0[v] : 0.f;
      float g1v = (H0s[m1 * HP + n] > (f16)0) ? c1[v] : 0.f;
      H0s[m0 * HP + n] = (f16)g0v;
      H0s[m1 * HP + n] = (f16)g1v;
    }
  }
  __syncthreads();

  // ---- g_x = g_z0 @ W0^T -> [32x64]; wave -> (col tile = wave&3, row tile = wave>>2)
  {
    int nt = wave & 3, mt = wave >> 2;
    int nb = nt * 16;
    const f16* gb = H0s + mt * 16 * HP;
    v8f c = {};
#pragma unroll
    for (int k = 0; k < 8; ++k)
      c = wmma32(frag_a(gb, HP, k * 32), frag_b(W0f + (size_t)nb * 256, 256, k * 32), c);
#pragma unroll
    for (int v = 0; v < 8; ++v)
      GXs[(mt * 16 + v + 8 * chi) * 64 + nb + cn] = c[v];
  }
  __syncthreads();

  // ---- chain through pos_enc derivative -> grad wrt points
  if (t < MT) {
    size_t s = s0 + t;
    const float* p = points + s * 3;
    float g0 = GXs[t * 64 + 0], g1 = GXs[t * 64 + 1], g2 = GXs[t * 64 + 2];
#pragma unroll
    for (int f = 0; f < 6; ++f) {
      float fr = PI_F * (float)(1 << f);
      float g[3];
#pragma unroll
      for (int d = 0; d < 3; ++d) {
        float xb = p[d] * fr;
        float gs = GXs[t * 64 + 3 + f * 6 + d];
        float gc = GXs[t * 64 + 3 + f * 6 + 3 + d];
        g[d] = fr * (__cosf(xb) * gs - __sinf(xb) * gc);
      }
      g0 += g[0]; g1 += g[1]; g2 += g[2];
    }
    gradp[s * 3 + 0] = g0; gradp[s * 3 + 1] = g1; gradp[s * 3 + 2] = g2;
  }
}

// ---------------- color MLP (WMMA layer0) + per-sample alpha math ----------------
// 64 samples/block, 8 waves; wave owns a 32-col slice, B frag reused for 4 row tiles.
__global__ __launch_bounds__(256) void k_color(
    const float* __restrict__ points, const float* __restrict__ dirs,
    const float* __restrict__ feats, const float* __restrict__ gradp,
    const float* __restrict__ t_st, const float* __restrict__ s_var,
    const f16* __restrict__ Wc0T, const float* __restrict__ bc0,
    const float* __restrict__ Wc1, const float* __restrict__ bc1,
    float* __restrict__ norml, float* __restrict__ rgb,
    float* __restrict__ aout, float* __restrict__ logaout, float* __restrict__ sdf_out) {
  const int MT = 64, CP = 40, HP = 272;
  __shared__ f16   Cs[64 * 40];
  __shared__ f16   Hs[64 * 272];
  __shared__ float Ns[64 * 3];
  int t = threadIdx.x;
  int s0 = blockIdx.x * MT;

  if (t < MT) {
    size_t s = s0 + t;
    float gx = gradp[s * 3], gy = gradp[s * 3 + 1], gz = gradp[s * 3 + 2];
    float inv = 1.0f / (sqrtf(gx * gx + gy * gy + gz * gz) + 1e-10f);
    float nx = gx * inv, ny = gy * inv, nz = gz * inv;
    Ns[t * 3] = nx; Ns[t * 3 + 1] = ny; Ns[t * 3 + 2] = nz;
    norml[s * 3] = nx; norml[s * 3 + 1] = ny; norml[s * 3 + 2] = nz;
  }
  __syncthreads();
  for (int idx = t; idx < MT * CP; idx += 256) {
    int r = idx / CP, c = idx % CP;
    size_t s = s0 + r;
    float v = 0.f;
    if (c < 3)       v = dirs[s * 3 + c];
    else if (c < 19) v = feats[s * 16 + (c - 3)];
    else if (c < 22) v = points[s * 3 + (c - 19)];
    else if (c < 25) v = Ns[r * 3 + (c - 22)];
    Cs[r * CP + c] = (f16)v;
  }
  __syncthreads();

  int wave = t >> 5;
  int lane = t & 31, cn = lane & 15, chi = lane >> 4;
  v16h a[4];
#pragma unroll
  for (int mt = 0; mt < 4; ++mt) a[mt] = frag_a(Cs + mt * 16 * CP, CP, 0);
#pragma unroll
  for (int nt = 0; nt < 2; ++nt) {
    int nb = wave * 32 + nt * 16;
    v16h b = frag_b(Wc0T + (size_t)nb * 32, 32, 0);
    float bb = bc0[nb + cn];
#pragma unroll
    for (int mt = 0; mt < 4; ++mt) {
      v8f c = {};
      c = wmma32(a[mt], b, c);
#pragma unroll
      for (int v = 0; v < 8; ++v)
        Hs[(mt * 16 + v + 8 * chi) * HP + nb + cn] = (f16)fmaxf(c[v] + bb, 0.f);
    }
  }
  __syncthreads();

  if (t < 192) {  // 256 -> 3 via VALU dot products, r=t&63, j=t>>6
    int r = t & 63, j = t >> 6;
    float accv = bc1[j];
    for (int k = 0; k < 256; ++k) accv += (float)Hs[r * HP + k] * Wc1[(size_t)k * 3 + j];
    rgb[(size_t)(s0 + r) * 3 + j] = 1.0f / (1.0f + __expf(-accv));
  }
  if (t < MT) {  // NeuS alpha math
    size_t s = s0 + t;
    float sdf = feats[s * 16];
    float dS = gradp[s * 3] * dirs[s * 3] + gradp[s * 3 + 1] * dirs[s * 3 + 1] +
               gradp[s * 3 + 2] * dirs[s * 3 + 2];
    dS = fminf(dS, 0.f);                               // -relu(-x)
    float inv_s = fminf(fmaxf(__expf(s_var[0]), 1e-6f), 1e6f);
    float cdf = 1.0f / (1.0f + __expf(-inv_s * sdf));
    float rho = fmaxf(inv_s * (cdf - 1.0f) * dS, 0.f);
    float alpha = 1.0f - __expf(-rho * STEP);
    float a_ = fminf(fmaxf(alpha, 0.f), 1.0f - 1e-7f);
    aout[s] = a_;
    logaout[s] = log1pf(-a_);
    sdf_out[s] = sdf;
  }
}

// ---------------- 3-phase scan of log-transmittance ----------------
__global__ __launch_bounds__(256) void k_scan1(const float* __restrict__ loga,
                                               float* __restrict__ incl, float* __restrict__ bsums) {
  __shared__ float sm[256];
  int t = threadIdx.x;
  int base = blockIdx.x * 4096 + t * 16;
  float loc[16]; float run = 0.f;
#pragma unroll
  for (int i = 0; i < 16; ++i) { run += loga[base + i]; loc[i] = run; }
  sm[t] = run; __syncthreads();
  float v = run;
  for (int off = 1; off < 256; off <<= 1) {
    float x = (t >= off) ? sm[t - off] : 0.f;
    __syncthreads();
    v += x; sm[t] = v;
    __syncthreads();
  }
  float excl = v - run;
#pragma unroll
  for (int i = 0; i < 16; ++i) incl[base + i] = excl + loc[i];
  if (t == 255) bsums[blockIdx.x] = v;
}

__global__ void k_scan2(float* __restrict__ bsums, float* __restrict__ boff) {
  __shared__ float sm[128];
  int t = threadIdx.x;
  float v = bsums[t]; sm[t] = v; __syncthreads();
  float tot = v;
  for (int off = 1; off < 128; off <<= 1) {
    float x = (t >= off) ? sm[t - off] : 0.f;
    __syncthreads();
    tot += x; sm[t] = tot;
    __syncthreads();
  }
  boff[t] = tot - v;  // exclusive
}

__global__ void k_scan3(const float* __restrict__ incl, const float* __restrict__ boff,
                        const float* __restrict__ loga, float* __restrict__ E) {
  int i = blockIdx.x * 256 + threadIdx.x;
  if (i >= N_SAMPLES) return;
  E[i] = incl[i] + boff[i >> 12] - loga[i];  // exclusive cumsum
}

// ---------------- segment starts, weights, per-ray reduction ----------------
__global__ void k_segstart(const int* __restrict__ ridx, const float* __restrict__ E,
                           float* __restrict__ segE, int* __restrict__ segI) {
  int i = blockIdx.x * 256 + threadIdx.x;
  if (i >= N_SAMPLES) return;
  if (i == 0 || ridx[i] != ridx[i - 1]) { segE[ridx[i]] = E[i]; segI[ridx[i]] = i; }
}

__global__ void k_weights(const int* __restrict__ ridx, const float* __restrict__ E,
                          const float* __restrict__ segE, const float* __restrict__ a,
                          float* __restrict__ w) {
  int i = blockIdx.x * 256 + threadIdx.x;
  if (i >= N_SAMPLES) return;
  w[i] = a[i] * __expf(E[i] - segE[ridx[i]]);
}

__global__ void k_ray(const int* __restrict__ segI, const int* __restrict__ ridx,
                      const float* __restrict__ w, const float* __restrict__ t_st,
                      const float* __restrict__ rgb, const float* __restrict__ norml,
                      float* __restrict__ color, float* __restrict__ pnorm,
                      float* __restrict__ opac, float* __restrict__ depth) {
  int r = blockIdx.x * 256 + threadIdx.x;
  if (r >= N_RAYS) return;
  float c0 = 0, c1 = 0, c2 = 0, n0 = 0, n1 = 0, n2 = 0, op = 0, dp = 0;
  int i = segI[r];
  if (i >= 0) {
    while (i < N_SAMPLES && ridx[i] == r) {
      float wi = w[i];
      op += wi;
      dp += wi * (t_st[i] + STEP * 0.5f);
      c0 += wi * rgb[(size_t)i * 3 + 0];
      c1 += wi * rgb[(size_t)i * 3 + 1];
      c2 += wi * rgb[(size_t)i * 3 + 2];
      n0 += wi * norml[(size_t)i * 3 + 0];
      n1 += wi * norml[(size_t)i * 3 + 1];
      n2 += wi * norml[(size_t)i * 3 + 2];
      ++i;
    }
  }
  color[r * 3] = c0; color[r * 3 + 1] = c1; color[r * 3 + 2] = c2;
  float inv = 1.0f / (sqrtf(n0 * n0 + n1 * n1 + n2 * n2) + 1e-10f);
  pnorm[r * 3] = n0 * inv; pnorm[r * 3 + 1] = n1 * inv; pnorm[r * 3 + 2] = n2 * inv;
  opac[r] = op; depth[r] = dp;
}

// ---------------- random SDF probe (1024 pts, negligible cost) ----------------
__global__ void k_random(const float* __restrict__ rp,
                         const float* __restrict__ Wg0, const float* __restrict__ bg0,
                         const float* __restrict__ Wg1, const float* __restrict__ bg1,
                         const float* __restrict__ Wg2, const float* __restrict__ bg2,
                         float* __restrict__ out) {
  int i = blockIdx.x * 64 + threadIdx.x;
  if (i >= 1024) return;
  float p[3] = {rp[i * 3], rp[i * 3 + 1], rp[i * 3 + 2]};
  float enc[39];
  enc[0] = p[0]; enc[1] = p[1]; enc[2] = p[2];
#pragma unroll
  for (int f = 0; f < 6; ++f) {
    float fr = PI_F * (float)(1 << f);
#pragma unroll
    for (int d = 0; d < 3; ++d) {
      enc[3 + f * 6 + d]     = __sinf(p[d] * fr);
      enc[3 + f * 6 + 3 + d] = __cosf(p[d] * fr);
    }
  }
  float h0[256], h1[256];
  for (int j = 0; j < 256; ++j) {
    float z = bg0[j];
    for (int k = 0; k < 39; ++k) z += enc[k] * Wg0[(size_t)k * 256 + j];
    h0[j] = fmaxf(z, 0.f);
  }
  for (int j = 0; j < 256; ++j) {
    float z = bg1[j];
    for (int k = 0; k < 256; ++k) z += h0[k] * Wg1[(size_t)k * 256 + j];
    h1[j] = fmaxf(z, 0.f);
  }
  float z = bg2[0];
  for (int k = 0; k < 256; ++k) z += h1[k] * Wg2[(size_t)k * 16];
  out[i] = z;
}

// ---------------- host orchestration ----------------
extern "C" void kernel_launch(void* const* d_in, const int* in_sizes, int n_in,
                              void* d_out, int out_size, void* d_ws, size_t ws_size,
                              hipStream_t stream) {
  const float* rays_o = (const float*)d_in[0];
  const float* rays_d = (const float*)d_in[1];
  const int*   ridx   = (const int*)d_in[2];
  const float* t_st   = (const float*)d_in[3];
  const float* rpts   = (const float*)d_in[4];
  const float* s_var  = (const float*)d_in[5];
  const float* Wg0 = (const float*)d_in[6];  const float* bg0 = (const float*)d_in[7];
  const float* Wg1 = (const float*)d_in[8];  const float* bg1 = (const float*)d_in[9];
  const float* Wg2 = (const float*)d_in[10]; const float* bg2 = (const float*)d_in[11];
  const float* Wc0 = (const float*)d_in[12]; const float* bc0 = (const float*)d_in[13];
  const float* Wc1 = (const float*)d_in[14]; const float* bc1 = (const float*)d_in[15];

  float* out = (float*)d_out;
  float* o_color = out;                 // 16384*3
  float* o_pnorm = out + 49152;         // 16384*3
  float* o_opac  = out + 98304;         // 16384
  float* o_depth = out + 114688;        // 16384
  float* o_w     = out + 131072;        // 524288
  float* o_sdf   = out + 655360;        // 524288
  float* o_rand  = out + 1179648;       // 1024

  char* ws = (char*)d_ws;
  auto alloc = [&](size_t bytes) -> void* {
    void* p = (void*)ws; ws += (bytes + 255) & ~(size_t)255; return p;
  };
  const size_t S = N_SAMPLES, R = N_RAYS;
  float* points = (float*)alloc(S * 3 * 4);
  float* dirs   = (float*)alloc(S * 3 * 4);
  float* feats  = (float*)alloc(S * 16 * 4);
  float* gradp  = (float*)alloc(S * 3 * 4);
  float* norml  = (float*)alloc(S * 3 * 4);
  float* rgb    = (float*)alloc(S * 3 * 4);
  float* abuf   = (float*)alloc(S * 4);
  float* lbuf   = (float*)alloc(S * 4);
  float* incl   = (float*)alloc(S * 4);
  float* Ebuf   = (float*)alloc(S * 4);
  float* bsums  = (float*)alloc(128 * 4);
  float* boff   = (float*)alloc(128 * 4);
  float* segE   = (float*)alloc(R * 4);
  int*   segI   = (int*)alloc(R * 4);
  f16* W0T  = (f16*)alloc(256 * 64 * 2);
  f16* W1T  = (f16*)alloc(65536 * 2);
  f16* W1f  = (f16*)alloc(65536 * 2);
  f16* W2T  = (f16*)alloc(16 * 256 * 2);
  f16* W0f  = (f16*)alloc(64 * 256 * 2);
  f16* Wc0T = (f16*)alloc(256 * 32 * 2);

  hipMemsetAsync(segI, 0xFF, R * 4, stream);  // -1 = empty ray

  k_wprep<<<256, 256, 0, stream>>>(Wg0, Wg1, Wg2, Wc0, W0T, W1T, W1f, W2T, W0f, Wc0T);
  k_prep<<<S / 256, 256, 0, stream>>>(rays_o, rays_d, ridx, t_st, points, dirs);
  k_geom<<<S / 32, 256, 0, stream>>>(points, W0T, bg0, W1T, bg1, W2T, bg2, W1f, W0f,
                                     Wg2, feats, gradp);
  k_color<<<S / 64, 256, 0, stream>>>(points, dirs, feats, gradp, t_st, s_var,
                                      Wc0T, bc0, Wc1, bc1, norml, rgb, abuf, lbuf, o_sdf);
  k_scan1<<<128, 256, 0, stream>>>(lbuf, incl, bsums);
  k_scan2<<<1, 128, 0, stream>>>(bsums, boff);
  k_scan3<<<S / 256, 256, 0, stream>>>(incl, boff, lbuf, Ebuf);
  k_segstart<<<S / 256, 256, 0, stream>>>(ridx, Ebuf, segE, segI);
  k_weights<<<S / 256, 256, 0, stream>>>(ridx, Ebuf, segE, abuf, o_w);
  k_ray<<<R / 256, 256, 0, stream>>>(segI, ridx, o_w, t_st, rgb, norml,
                                     o_color, o_pnorm, o_opac, o_depth);
  k_random<<<16, 64, 0, stream>>>(rpts, Wg0, bg0, Wg1, bg1, Wg2, bg2, o_rand);
}
